// Gpt2Block_59828894433635
// MI455X (gfx1250) — matrix operand order repaired and checked by reading, not compiled, not run on hardware
//
#include <hip/hip_runtime.h>
#include <hip/hip_bf16.h>

// ---------- types ----------
typedef __attribute__((ext_vector_type(16))) __bf16          v16bf;
typedef __attribute__((ext_vector_type(8)))  float           v8f;
typedef __attribute__((ext_vector_type(8)))  unsigned short  ush8;
typedef __attribute__((ext_vector_type(16))) unsigned short  ush16;
typedef __attribute__((ext_vector_type(4)))  unsigned int    u32x4;
typedef __attribute__((ext_vector_type(8)))  int             i32x8;
typedef __attribute__((ext_vector_type(4)))  int             i32x4;

union U16frag { ush16 v; struct { ush8 lo; ush8 hi; } p; };

// ---------- helpers ----------
__device__ inline unsigned short f2bf(float f) {
    union { float f; unsigned u; } c; c.f = f;
    unsigned u = c.u;
    unsigned r = u + 0x7FFFu + ((u >> 16) & 1u);   // round-to-nearest-even
    return (unsigned short)(r >> 16);
}
__device__ inline float bf2f(unsigned short u) {
    union { unsigned u; float f; } c; c.u = ((unsigned)u) << 16; return c.f;
}
__device__ inline v8f zero8() {
    v8f z = {0.f,0.f,0.f,0.f,0.f,0.f,0.f,0.f}; return z;
}
__device__ inline v8f wmma_bf16(ush16 a, ush16 b, v8f c) {
    return __builtin_amdgcn_wmma_f32_16x16x32_bf16(
        /*neg_a=*/false, __builtin_bit_cast(v16bf, a),
        /*neg_b=*/false, __builtin_bit_cast(v16bf, b),
        /*c_mod=*/(short)0, c, /*reuse_a=*/false, /*reuse_b=*/false);
}
__device__ inline float gelu_tanh(float x) {
    float x3 = x * x * x;
    return 0.5f * x * (1.f + tanhf(0.7978845608028654f * (x + 0.044715f * x3)));
}

// async global->LDS 16-byte copy (GVS mode: uniform 64-bit base + 32-bit lane offset)
__device__ inline void async_copy_b128(unsigned lds_byte_addr, const void* sbase,
                                       unsigned voff_bytes) {
    asm volatile("global_load_async_to_lds_b128 %0, %1, %2"
                 :: "v"(lds_byte_addr), "v"(voff_bytes), "s"(sbase) : "memory");
}
__device__ inline void async_wait0() {
    asm volatile("s_wait_asynccnt 0x0" ::: "memory");
}

// ---------- TDM: 2D bf16 tile load, 256B rows + 16B LDS pad (->136-elem rows) ----------
__device__ inline void tdm_load_2d_bf16(unsigned lds_addr, const void* gptr,
                                        unsigned tile_d0, unsigned tile_d1,
                                        unsigned tensor_d0, unsigned tensor_d1,
                                        unsigned stride_elems)
{
    unsigned long long ga = (unsigned long long)(size_t)gptr;
    u32x4 g0;
    g0[0] = 1u;                                            // count=1 (valid D#)
    g0[1] = lds_addr;                                      // LDS byte address
    g0[2] = (unsigned)(ga & 0xFFFFFFFFu);                  // global_addr[31:0]
    g0[3] = (unsigned)((ga >> 32) & 0x01FFFFFFu) | (2u << 30); // addr[56:32] | type=2
    i32x8 g1;
    // data_size=1 (2B) | pad_enable | pad_interval=5 (64 DW = 256B) | pad_amount=3 (4 DW = 16B)
    g1[0] = (int)((1u << 16) | (1u << 20) | (5u << 22) | (3u << 25));
    g1[1] = (int)((tensor_d0 & 0xFFFFu) << 16);            // tensor_dim0[15:0]
    g1[2] = (int)(((tensor_d0 >> 16) & 0xFFFFu) | ((tensor_d1 & 0xFFFFu) << 16));
    g1[3] = (int)(((tensor_d1 >> 16) & 0xFFFFu) | ((tile_d0 & 0xFFFFu) << 16));
    g1[4] = (int)(tile_d1 & 0xFFFFu);                      // tile_dim1 | tile_dim2=0
    g1[5] = (int)stride_elems;                             // tensor_dim0_stride[31:0]
    g1[6] = 0;
    g1[7] = 0;
    i32x4 gz = {0, 0, 0, 0};
#if __has_include(<hip/amd_detail/amd_gfx1250_TDM.h>)
    i32x8 gz8 = {0, 0, 0, 0, 0, 0, 0, 0};
    __builtin_amdgcn_tensor_load_to_lds(g0, g1, gz, gz, gz8, 0);
#else
    __builtin_amdgcn_tensor_load_to_lds(g0, g1, gz, gz, 0);
#endif
}

__device__ inline float blockReduceSum256(float v) {
    __shared__ float tmp[8];
    #pragma unroll
    for (int off = 16; off > 0; off >>= 1) v += __shfl_xor(v, off, 32);
    int wv = threadIdx.x >> 5;
    __syncthreads();
    if ((threadIdx.x & 31) == 0) tmp[wv] = v;
    __syncthreads();
    float r = tmp[0];
    #pragma unroll
    for (int i = 1; i < 8; i++) r += tmp[i];
    return r;
}

// ---------- LayerNorm: f32 in -> bf16 out ----------
__global__ __launch_bounds__(256) void ln_kernel(
    const float* __restrict__ x, const float* __restrict__ w,
    const float* __restrict__ bia, unsigned short* __restrict__ out, int E)
{
    int row = blockIdx.x;
    const float* xr = x + (size_t)row * E;
    int tid = threadIdx.x;
    float s = 0.f;
    for (int i = tid; i < E; i += 256) s += xr[i];
    s = blockReduceSum256(s);
    float mu = s / (float)E;
    float var = 0.f;
    for (int i = tid; i < E; i += 256) { float d = xr[i] - mu; var += d * d; }
    var = blockReduceSum256(var);
    float rs = rsqrtf(var / (float)E + 1e-5f);
    for (int i = tid; i < E; i += 256) {
        float y = (xr[i] - mu) * rs * w[i] + bia[i];
        out[(size_t)row * E + i] = f2bf(y);
    }
}

// ---------- transpose-convert: out[n*K+k] = bf16(in[k*N+n]) ----------
__global__ __launch_bounds__(256) void transpose_bf16_kernel(
    const float* __restrict__ in, unsigned short* __restrict__ out, int K, int N)
{
    __shared__ float tile[32][33];
    int k0 = blockIdx.x * 32, n0 = blockIdx.y * 32;
    int tx = threadIdx.x & 31, ty = threadIdx.x >> 5;
    for (int i = ty; i < 32; i += 8)
        tile[i][tx] = in[(size_t)(k0 + i) * N + (n0 + tx)];
    __syncthreads();
    for (int i = ty; i < 32; i += 8)
        out[(size_t)(n0 + i) * K + (k0 + tx)] = f2bf(tile[tx][i]);
}

// ---------- WMMA GEMM: C[M,N] = A[M,K](bf16) * Bt[N,K](bf16) + bias ----------
// B tile (64 cols x 256 k) staged in LDS via async-to-LDS; shared by all 8 waves.
// MODE 0: out bf16   MODE 1: out f32 = res + acc + bias   MODE 2: bf16 = gelu(acc+bias)
template <int MODE>
__global__ __launch_bounds__(256) void gemm_kernel(
    const unsigned short* __restrict__ A, const unsigned short* __restrict__ Bt,
    const float* __restrict__ bias, const float* __restrict__ res,
    void* __restrict__ outp, int M, int N, int K)
{
    __shared__ alignas(16) unsigned short Bs[64][264];   // 256 k + 8 pad

    int w    = threadIdx.x >> 5;
    int lane = threadIdx.x & 31;
    int half = lane >> 4;
    int lm   = lane & 15;
    int tile_m = blockIdx.y * 8 + w;
    int n0     = blockIdx.x * 64;
    int row    = tile_m * 16 + lm;

    const unsigned short* arow  = A + (size_t)row * K + half * 16;
    const unsigned short* gbase = Bt + (size_t)n0 * K;   // uniform base for async copies

    // async fill mapping: 4 threads per B-column, 8 x b128 chunks each
    int tcol = threadIdx.x >> 2;   // 0..63
    int tseg = threadIdx.x & 3;    // 0..3 (64 elems each)

    v8f acc0 = zero8(), acc1 = zero8(), acc2 = zero8(), acc3 = zero8();

    for (int kk0 = 0; kk0 < K; kk0 += 256) {
        __syncthreads();   // previous tile fully consumed
        #pragma unroll
        for (int c = 0; c < 8; c++) {
            int elem = tseg * 64 + c * 8;
            unsigned voff = (unsigned)(((unsigned)tcol * (unsigned)K
                                        + (unsigned)(kk0 + elem)) * 2u);
            async_copy_b128((unsigned)(size_t)&Bs[tcol][elem], gbase, voff);
        }
        async_wait0();
        __syncthreads();

        #pragma unroll
        for (int k8 = 0; k8 < 8; k8++) {
            int kk = kk0 + k8 * 32;
            __builtin_prefetch(arow + kk + 256, 0, 0);
            U16frag a;
            a.p.lo = *(const ush8*)(arow + kk);
            a.p.hi = *(const ush8*)(arow + kk + 8);
            int ko = k8 * 32 + half * 16;
            ush16 b0 = *(const ush16*)(&Bs[0 * 16 + lm][ko]);
            ush16 b1 = *(const ush16*)(&Bs[1 * 16 + lm][ko]);
            ush16 b2 = *(const ush16*)(&Bs[2 * 16 + lm][ko]);
            ush16 b3 = *(const ush16*)(&Bs[3 * 16 + lm][ko]);
            acc0 = wmma_bf16(a.v, b0, acc0);
            acc1 = wmma_bf16(a.v, b1, acc1);
            acc2 = wmma_bf16(a.v, b2, acc2);
            acc3 = wmma_bf16(a.v, b3, acc3);
        }
    }

    v8f accs[4] = {acc0, acc1, acc2, acc3};
    #pragma unroll
    for (int t = 0; t < 4; t++) {
        int col = n0 + t * 16 + lm;
        float bv = bias[col];
        #pragma unroll
        for (int j = 0; j < 8; j++) {
            int r = tile_m * 16 + half * 8 + j;
            size_t idx = (size_t)r * N + col;
            float v = accs[t][j] + bv;
            if (MODE == 0)      ((unsigned short*)outp)[idx] = f2bf(v);
            else if (MODE == 1) ((float*)outp)[idx] = res[idx] + v;
            else                ((unsigned short*)outp)[idx] = f2bf(gelu_tanh(v));
        }
    }
}

// ---------- flash attention ----------
// qkv: bf16 [B,S,3,H,D]  (offset = t*2048 + h*128 + d),  ctx: bf16 [B,S,E]
__global__ __launch_bounds__(256) void attn_kernel(
    const unsigned short* __restrict__ qkv, unsigned short* __restrict__ ctx)
{
    const int S = 2048, E = 2048, Dh = 128, HD3 = 6144;
    int i0 = blockIdx.x * 128;   // query block start
    int h  = blockIdx.y;
    int b  = blockIdx.z;
    int tid  = threadIdx.x;
    int w    = tid >> 5;
    int lane = tid & 31;
    int half = lane >> 4;
    int lm   = lane & 15;

    __shared__ alignas(16) unsigned short Qs[128][136];
    __shared__ alignas(16) unsigned short Ks[64][136];
    __shared__ alignas(16) unsigned short Vt[128][72];   // transposed: [d][key]
    __shared__ alignas(16) unsigned short Ps[8][16][72];

    const float scale = 0.08838834764831845f;  // rsqrt(128)
    size_t base = (size_t)(b * S) * HD3;

    // load Q (pre-scaled)
    for (int idx = tid; idx < 128 * 128; idx += 256) {
        int r = idx >> 7, d = idx & 127;
        unsigned short u = qkv[base + (size_t)(i0 + r) * HD3 + 0 * E + h * Dh + d];
        Qs[r][d] = f2bf(bf2f(u) * scale);
    }

    float mrow[8], lrow[8];
    v8f o[8];
    #pragma unroll
    for (int j = 0; j < 8; j++) { mrow[j] = -1e30f; lrow[j] = 0.f; o[j] = zero8(); }

    for (int j0 = 0; j0 < i0 + 128; j0 += 64) {
        __syncthreads();   // previous K/V tiles fully consumed

        // K tile via Tensor Data Mover (one wave issues; overlapped with V fill)
        if (tid < 32) {
            const unsigned short* gk = qkv + base + (size_t)j0 * HD3 + 1 * E + h * Dh;
            tdm_load_2d_bf16((unsigned)(size_t)&Ks[0][0], gk,
                             /*tile*/128, 64, /*tensor*/128, 64, /*stride*/HD3);
        }
        // V tile, stored transposed for contiguous PV B-fragments
        for (int idx = tid; idx < 64 * 128; idx += 256) {
            int r = idx >> 7, d = idx & 127;
            Vt[d][r] = qkv[base + (size_t)(j0 + r) * HD3 + 2 * E + h * Dh + d];
        }
        if (tid < 32) __builtin_amdgcn_s_wait_tensorcnt(0);
        __syncthreads();

        if (j0 <= i0 + w * 16 + 15) {   // wave-uniform: skip fully-masked key blocks
            // ---- S = Q * K^T (16 rows x 64 keys) ----
            v8f st[4];
            #pragma unroll
            for (int t = 0; t < 4; t++) st[t] = zero8();
            #pragma unroll
            for (int ks = 0; ks < 4; ks++) {
                int kk = ks * 32;
                U16frag a;
                const unsigned short* qrow = &Qs[w * 16 + lm][kk + half * 16];
                a.p.lo = *(const ush8*)qrow;
                a.p.hi = *(const ush8*)(qrow + 8);
                #pragma unroll
                for (int t = 0; t < 4; t++) {
                    ush16 bq = *(const ush16*)(&Ks[t * 16 + lm][kk + half * 16]);
                    st[t] = wmma_bf16(a.v, bq, st[t]);
                }
            }
            // ---- online softmax (each wave owns its 16 rows) ----
            #pragma unroll
            for (int j = 0; j < 8; j++) {
                int rg = i0 + w * 16 + half * 8 + j;
                float sv[4];
                float mj = -1e30f;
                #pragma unroll
                for (int t = 0; t < 4; t++) {
                    int kg = j0 + t * 16 + lm;
                    float s = st[t][j];
                    if (kg > rg) s = -1e9f;
                    sv[t] = s;
                    mj = fmaxf(mj, s);
                }
                #pragma unroll
                for (int off = 8; off >= 1; off >>= 1)
                    mj = fmaxf(mj, __shfl_xor(mj, off, 32));
                float nm    = fmaxf(mrow[j], mj);
                float alpha = __expf(mrow[j] - nm);
                float ps = 0.f;
                #pragma unroll
                for (int t = 0; t < 4; t++) {
                    float p = __expf(sv[t] - nm);
                    ps += p;
                    Ps[w][half * 8 + j][t * 16 + lm] = f2bf(p);
                }
                #pragma unroll
                for (int off = 8; off >= 1; off >>= 1)
                    ps += __shfl_xor(ps, off, 32);
                lrow[j] = lrow[j] * alpha + ps;
                mrow[j] = nm;
                #pragma unroll
                for (int td = 0; td < 8; td++) o[td][j] *= alpha;
            }
            // ---- O += P * V ----
            #pragma unroll
            for (int ks = 0; ks < 2; ks++) {
                int kk = ks * 32;
                U16frag a;
                const unsigned short* prow = &Ps[w][lm][kk + half * 16];
                a.p.lo = *(const ush8*)prow;
                a.p.hi = *(const ush8*)(prow + 8);
                #pragma unroll
                for (int td = 0; td < 8; td++) {
                    ush16 bv = *(const ush16*)(&Vt[td * 16 + lm][kk + half * 16]);
                    o[td] = wmma_bf16(a.v, bv, o[td]);
                }
            }
        }
    }

    // ---- normalize + write ctx ----
    #pragma unroll
    for (int td = 0; td < 8; td++) {
        #pragma unroll
        for (int j = 0; j < 8; j++) {
            int rg = i0 + w * 16 + half * 8 + j;
            float v = o[td][j] / lrow[j];
            ctx[(size_t)(b * S + rg) * E + h * Dh + td * 16 + lm] = f2bf(v);
        }
    }
}

// ---------- host launch ----------
extern "C" void kernel_launch(void* const* d_in, const int* in_sizes, int n_in,
                              void* d_out, int out_size, void* d_ws, size_t ws_size,
                              hipStream_t stream)
{
    const int Bn = 4, S = 2048, E = 2048, F = 8192, N3 = 6144;
    const int M = Bn * S;  // 8192

    const float* x        = (const float*)d_in[0];
    const float* ln1_w    = (const float*)d_in[1];
    const float* ln1_b    = (const float*)d_in[2];
    const float* c_attn_w = (const float*)d_in[3];   // [2048, 6144]
    const float* c_attn_b = (const float*)d_in[4];   // [6144]
    const float* c_proj_w = (const float*)d_in[5];   // [2048, 2048]
    const float* c_proj_b = (const float*)d_in[6];
    const float* ln2_w    = (const float*)d_in[7];
    const float* ln2_b    = (const float*)d_in[8];
    const float* c_fc_w   = (const float*)d_in[9];   // [2048, 8192]
    const float* c_fc_b   = (const float*)d_in[10];
    const float* c_mp_w   = (const float*)d_in[11];  // [8192, 2048]
    const float* c_mp_b   = (const float*)d_in[12];
    float* out = (float*)d_out;

    size_t off = 0;
    auto take = [&](size_t bytes) -> void* {
        void* r = (char*)d_ws + off;
        off += (bytes + 255) & ~(size_t)255;
        return r;
    };
    unsigned short* h1     = (unsigned short*)take((size_t)M * E * 2);
    unsigned short* wqkvT  = (unsigned short*)take((size_t)N3 * E * 2);
    unsigned short* qkv    = (unsigned short*)take((size_t)M * N3 * 2);
    unsigned short* ctx    = (unsigned short*)take((size_t)M * E * 2);
    unsigned short* wprojT = (unsigned short*)take((size_t)E * E * 2);
    float*          x1     = (float*)take((size_t)M * E * 4);
    unsigned short* h2     = (unsigned short*)take((size_t)M * E * 2);
    unsigned short* wfcT   = (unsigned short*)take((size_t)F * E * 2);
    unsigned short* fca    = (unsigned short*)take((size_t)M * F * 2);
    unsigned short* wmlpT  = (unsigned short*)take((size_t)E * F * 2);

    dim3 blk(256);

    // weight conversions (f32 -> bf16, transposed to [N,K])
    transpose_bf16_kernel<<<dim3(E / 32, N3 / 32), blk, 0, stream>>>(c_attn_w, wqkvT, E, N3);
    transpose_bf16_kernel<<<dim3(E / 32, E / 32),  blk, 0, stream>>>(c_proj_w, wprojT, E, E);
    transpose_bf16_kernel<<<dim3(E / 32, F / 32),  blk, 0, stream>>>(c_fc_w,   wfcT,   E, F);
    transpose_bf16_kernel<<<dim3(F / 32, E / 32),  blk, 0, stream>>>(c_mp_w,   wmlpT,  F, E);

    // LN1
    ln_kernel<<<dim3(M), blk, 0, stream>>>(x, ln1_w, ln1_b, h1, E);

    // QKV = h1 @ Wqkv + b  (bf16 out)
    gemm_kernel<0><<<dim3(N3 / 64, M / 128), blk, 0, stream>>>(
        h1, wqkvT, c_attn_b, nullptr, qkv, M, N3, E);

    // flash attention -> ctx (bf16)
    attn_kernel<<<dim3(S / 128, 16, Bn), blk, 0, stream>>>(qkv, ctx);

    // x1 = x + ctx @ Wproj + b  (f32 out)
    gemm_kernel<1><<<dim3(E / 64, M / 128), blk, 0, stream>>>(
        ctx, wprojT, c_proj_b, x, x1, M, E, E);

    // LN2
    ln_kernel<<<dim3(M), blk, 0, stream>>>(x1, ln2_w, ln2_b, h2, E);

    // fca = gelu(h2 @ Wfc + b)  (bf16 out)
    gemm_kernel<2><<<dim3(F / 64, M / 128), blk, 0, stream>>>(
        h2, wfcT, c_fc_b, nullptr, fca, M, F, E);

    // out = x1 + fca @ Wmlp + b  (f32 out)
    gemm_kernel<1><<<dim3(E / 64, M / 128), blk, 0, stream>>>(
        fca, wmlpT, c_mp_b, x1, out, M, E, F);
}
